// Emb_57930518889074
// MI455X (gfx1250) — matrix operand order repaired and checked by reading, not compiled, not run on hardware
//
#include <hip/hip_runtime.h>
#include <hip/hip_bf16.h>

// ---------------------------------------------------------------------------
// Problem constants
// ---------------------------------------------------------------------------
#define SEQ_LEN   336
#define D_MODEL   512
#define N_TOK     32768          // B*V = 64*512
#define N_EXPERT  6
#define FREQ_LEN  169            // 336/2 + 1
#define RDIM      505            // 336 + 169
#define DFT_KP    352            // 336 padded to mult of 32
#define DFT_NC    352            // 169 cos | pad | 169 sin | pad
#define RF_KP     512            // 505 (+1 bias col) padded
#define GATE_H    128

// Expert-kernel LDS layout (dynamic shared; ISA allows up to 320KB/WG)
#define XS_W      440            // u16 stride, >= (P-1)*step + K1 (max 432)
#define HS_W      560            // u16 stride, >= max Kp (544), mult of 8
#define HF_W      512            // f32 stride for patch-GEMM scratch
#define SM_XS     0
#define SM_HS     (32 * XS_W * 2)                 // 28160
#define SM_HF     (SM_HS + 32 * HS_W * 2)         // 64000
#define SM_TOTAL  (SM_HF + 32 * HF_W * 4)         // 129536 bytes

typedef __attribute__((ext_vector_type(16))) __bf16 v16bf;
typedef __attribute__((ext_vector_type(8)))  float  v8f;

union Frag16 { v16bf v; unsigned int d[8]; uint4 q[2]; };

// ---------------------------------------------------------------------------
// Helpers
// ---------------------------------------------------------------------------
__device__ __forceinline__ unsigned short f2bf(float f) {
  unsigned int u = __float_as_uint(f);
  unsigned int r = u + 0x7FFFu + ((u >> 16) & 1u);   // round-to-nearest-even
  return (unsigned short)(r >> 16);
}
__device__ __forceinline__ float bf2f(unsigned short h) {
  return __uint_as_float(((unsigned int)h) << 16);
}
__device__ __forceinline__ float gelu_exact(float v) {
  return 0.5f * v * (1.0f + erff(v * 0.70710678118654752f));
}
__device__ __forceinline__ v8f wmma_bf16(v16bf a, v16bf b, v8f c) {
  return __builtin_amdgcn_wmma_f32_16x16x32_bf16(
      /*neg_a=*/false, a, /*neg_b=*/false, b,
      /*c_mod=*/(short)0, c, /*reuse_a=*/false, /*reuse_b=*/false);
}

// ---------------------------------------------------------------------------
// k1: x (f32 [N,336]) -> bf16 [N,352] (zero padded)
// ---------------------------------------------------------------------------
__global__ __launch_bounds__(256) void k_x_to_bf16(const float* __restrict__ x,
                                                   unsigned short* __restrict__ Xbf) {
  int idx = blockIdx.x * 256 + threadIdx.x;
  if (idx >= N_TOK * DFT_KP) return;
  int col = idx % DFT_KP;
  int n   = idx / DFT_KP;
  float v = (col < SEQ_LEN) ? x[(size_t)n * SEQ_LEN + col] : 0.0f;
  Xbf[idx] = f2bf(v);
}

// ---------------------------------------------------------------------------
// B-tile layout (v_wmma_f32_16x16x32_bf16 B operand, wave32):
//   tile (tk,tn) of [Kp x Nc] stored at (tn*Kt + tk)*512 elements; lane l owns
//   16 contiguous bf16 at l*16: n = tn*16 + (l&15),
//                               k = tk*32 + ((l&16)?8:0) + ((v&4)?16:0) + 2*(v&3) + lo
// ---------------------------------------------------------------------------
__global__ __launch_bounds__(256) void k_pack_b_tiles(const float* __restrict__ W,
                                                      const float* __restrict__ biasRow,
                                                      int Krows, int Ncols, int KbRow,
                                                      int Kp, int Nc,
                                                      unsigned short* __restrict__ dst) {
  int idx = blockIdx.x * 256 + threadIdx.x;
  if (idx >= Kp * Nc) return;
  int lo   = idx & 1;
  int v    = (idx >> 1) & 7;
  int lane = (idx >> 4) & 31;
  int tile = idx >> 9;
  int Kt = Kp >> 5;
  int tk = tile % Kt;
  int tn = tile / Kt;
  int n = tn * 16 + (lane & 15);
  int k = tk * 32 + ((lane & 16) ? 8 : 0) + ((v & 4) ? 16 : 0) + 2 * (v & 3) + lo;
  float val = 0.0f;
  if (n < Ncols) {
    if (k < Krows)                  val = W[(size_t)k * Ncols + n];
    else if (k == KbRow && biasRow) val = biasRow[n];
  }
  dst[idx] = f2bf(val);
}

// DFT basis tiles: Bas[k,n] = cos(2*pi*k*f/336) for n=f<169; sin at n=176+f.
__global__ __launch_bounds__(256) void k_pack_dft_tiles(unsigned short* __restrict__ dst) {
  int idx = blockIdx.x * 256 + threadIdx.x;
  if (idx >= DFT_KP * DFT_NC) return;
  int lo   = idx & 1;
  int v    = (idx >> 1) & 7;
  int lane = (idx >> 4) & 31;
  int tile = idx >> 9;
  int Kt = DFT_KP >> 5;
  int tk = tile % Kt;
  int tn = tile / Kt;
  int n = tn * 16 + (lane & 15);
  int k = tk * 32 + ((lane & 16) ? 8 : 0) + ((v & 4) ? 16 : 0) + 2 * (v & 3) + lo;
  const float OM = 6.28318530717958647f / (float)SEQ_LEN;
  float val = 0.0f;
  if (k < SEQ_LEN) {
    if (n < FREQ_LEN)                         val = cosf((float)(k * n) * OM);
    else if (n >= 176 && n < 176 + FREQ_LEN)  val = sinf((float)(k * (n - 176)) * OM);
  }
  dst[idx] = f2bf(val);
}

// ---------------------------------------------------------------------------
// Generic wave-per-tile GEMM: C[M,Nc] f32 = A[M,Ka] bf16 (row-major) @ Bt tiles
// ---------------------------------------------------------------------------
__global__ __launch_bounds__(256) void k_gemm_wave(const unsigned short* __restrict__ A,
                                                   const unsigned short* __restrict__ Bt,
                                                   float* __restrict__ C,
                                                   int M, int Ka, int Nc) {
  int lane = threadIdx.x & 31;
  int wg   = blockIdx.x * 8 + (threadIdx.x >> 5);
  int Mt = M >> 4, Nt = Nc >> 4;
  if (wg >= Mt * Nt) return;
  int tm = wg % Mt;
  int tn = wg / Mt;
  int Kt = Ka >> 5;
  int row = tm * 16 + (lane & 15);
  int kh  = (lane & 16) ? 8 : 0;

  v8f acc = {0.f, 0.f, 0.f, 0.f, 0.f, 0.f, 0.f, 0.f};
  for (int tk = 0; tk < Kt; ++tk) {
    Frag16 a, b;
    const uint4* ap = (const uint4*)(A + (size_t)row * Ka + tk * 32 + kh);
    a.q[0] = ap[0];            // k .. k+7
    a.q[1] = ap[2];            // k+16 .. k+23
    const uint4* bp = (const uint4*)(Bt + ((size_t)(tn * Kt + tk)) * 512 + lane * 16);
    b.q[0] = bp[0];
    b.q[1] = bp[1];
    // speculative prefetch of next B tile (past-end prefetch is dropped by HW)
    __builtin_prefetch(Bt + ((size_t)(tn * Kt + tk + 1)) * 512 + lane * 16, 0, 1);
    acc = wmma_bf16(a.v, b.v, acc);
  }
  int rowhi = (lane & 16) ? 8 : 0;
#pragma unroll
  for (int r = 0; r < 8; ++r) {
    C[(size_t)(tm * 16 + rowhi + r) * Nc + tn * 16 + (lane & 15)] = acc[r];
  }
}

// ---------------------------------------------------------------------------
// k4: magnitude + rf assembly: rf bf16 [N,512] = [x | |rfft(x)| | 1.0 | 0]
// ---------------------------------------------------------------------------
__global__ __launch_bounds__(256) void k_build_rf(const float* __restrict__ x,
                                                  const float* __restrict__ RI,
                                                  unsigned short* __restrict__ rf) {
  int idx = blockIdx.x * 256 + threadIdx.x;
  if (idx >= N_TOK * RF_KP) return;
  int col = idx & (RF_KP - 1);
  int n   = idx >> 9;
  float v;
  if (col < SEQ_LEN) {
    v = x[(size_t)n * SEQ_LEN + col];
  } else if (col < RDIM) {
    int f = col - SEQ_LEN;
    float re = RI[(size_t)n * DFT_NC + f];
    float im = RI[(size_t)n * DFT_NC + 176 + f];
    v = sqrtf(re * re + im * im);
  } else if (col == RDIM) {
    v = 1.0f;                          // bias column for gate W1's b1 row
  } else {
    v = 0.0f;
  }
  rf[idx] = f2bf(v);
}

// ---------------------------------------------------------------------------
// k6: gate epilogue — GELU(h), 128x6 logits, top-4 softmax -> dense gates[N,6]
// ---------------------------------------------------------------------------
__global__ __launch_bounds__(256) void k_gate_topk(const float* __restrict__ Hg,
                                                   const float* __restrict__ W2g,
                                                   const float* __restrict__ b2g,
                                                   float* __restrict__ gates) {
  int n = blockIdx.x * 256 + threadIdx.x;
  if (n >= N_TOK) return;
  float lg[N_EXPERT];
#pragma unroll
  for (int e = 0; e < N_EXPERT; ++e) lg[e] = b2g[e];
  for (int j = 0; j < GATE_H; ++j) {
    float h = gelu_exact(Hg[(size_t)n * GATE_H + j]);
#pragma unroll
    for (int e = 0; e < N_EXPERT; ++e) lg[e] += h * W2g[j * N_EXPERT + e];
  }
  int sel[4]; float sv[4]; bool used[N_EXPERT] = {false, false, false, false, false, false};
#pragma unroll
  for (int t = 0; t < 4; ++t) {
    int best = 0; float bv = -3.4e38f;
#pragma unroll
    for (int e = 0; e < N_EXPERT; ++e)
      if (!used[e] && lg[e] > bv) { bv = lg[e]; best = e; }
    used[best] = true; sel[t] = best; sv[t] = bv;
  }
  float mx = sv[0], s = 0.f;
#pragma unroll
  for (int t = 0; t < 4; ++t) { sv[t] = expf(sv[t] - mx); s += sv[t]; }
  float og[N_EXPERT] = {0.f, 0.f, 0.f, 0.f, 0.f, 0.f};
#pragma unroll
  for (int t = 0; t < 4; ++t) og[sel[t]] = sv[t] / s;
#pragma unroll
  for (int e = 0; e < N_EXPERT; ++e) gates[(size_t)n * N_EXPERT + e] = og[e];
}

// ---------------------------------------------------------------------------
// Fused expert kernel: 32 tokens per workgroup. Dynamic LDS:
//   xs [32][440] bf16 : x rows with left-pad folded in (zeros outside data)
//   hs [32][560] bf16 : H rows (layer-1 output, gate-scaled, bias col at Ke)
//   Hf [32][512] f32  : patch-GEMM scratch (chunked over patches)
// Phase 1a (pl>=12): per-patch WMMA GEMM  Hf = xs_window @ W1tiles
// Phase 1b: GELU(Hf + b1) * gate -> hs   (VALU fallback for pl=6,3)
// Phase 2 : 8-wave WMMA GEMM  out[32,512] (+)= hs[32,Kp] @ W2tiles[Kp,512]
// ---------------------------------------------------------------------------
__global__ __launch_bounds__(256) void k_expert_fused(const float* __restrict__ x,
                                                      const float* __restrict__ W1,
                                                      const float* __restrict__ b1,
                                                      const unsigned short* __restrict__ W1t,
                                                      const unsigned short* __restrict__ W2t,
                                                      const float* __restrict__ gates,
                                                      float* __restrict__ out,
                                                      int pl, int step, int padl,
                                                      int inner, int P, int Ke, int Kp,
                                                      int patchWmma, int KT1, int NT1,
                                                      int innerPad, int CHP,
                                                      int eIdx, int accumulate) {
  extern __shared__ char smem[];
  unsigned short* xs = (unsigned short*)(smem + SM_XS);
  unsigned short* hb = (unsigned short*)(smem + SM_HS);
  float*          Hf = (float*)(smem + SM_HF);
#define XS(m, j) xs[(m) * XS_W + (j)]
#define HS(m, j) hb[(m) * HS_W + (j)]
#define HF(m, j) Hf[(m) * HF_W + (j)]

  const int tid = threadIdx.x;
  const size_t n0 = (size_t)blockIdx.x * 32;
  const int lane = tid & 31;
  const int w    = tid >> 5;
  const int kh   = (lane & 16) ? 8 : 0;
  const int mrow = lane & 15;
  const int rowhi = (lane & 16) ? 8 : 0;
  const v8f zero = {0.f, 0.f, 0.f, 0.f, 0.f, 0.f, 0.f, 0.f};

  // Phase 0: stage x rows (pad regions zeroed; width covers K-padded windows)
  for (int i2 = tid; i2 < 32 * XS_W; i2 += 256) {
    int m = i2 / XS_W, j = i2 % XS_W;
    int t = j - padl;
    float v = (t >= 0 && t < SEQ_LEN) ? x[(n0 + m) * SEQ_LEN + t] : 0.0f;
    XS(m, j) = f2bf(v);
  }
  __syncthreads();

  if (patchWmma) {
    // Phase 1a/1b: chunked over patches so Hf fits 32x512 f32
    int nchunk = (P + CHP - 1) / CHP;
    for (int ch = 0; ch < nchunk; ++ch) {
      int p0 = ch * CHP;
      int pc = (P - p0 < CHP) ? (P - p0) : CHP;
      int T = 2 * pc * NT1;
      for (int t = w; t < T; t += 8) {
        int mt = t & 1;
        int rest = t >> 1;
        int pp = rest / NT1;
        int nt = rest - pp * NT1;
        int p = p0 + pp;
        int row = mt * 16 + mrow;
        v8f acc = zero;
        for (int tk = 0; tk < KT1; ++tk) {
          Frag16 a, b;
          const unsigned int* apd =
              (const unsigned int*)&XS(row, 0) + ((p * step + tk * 32 + kh) >> 1);
          a.d[0] = apd[0]; a.d[1] = apd[1]; a.d[2] = apd[2]; a.d[3] = apd[3];
          a.d[4] = apd[8]; a.d[5] = apd[9]; a.d[6] = apd[10]; a.d[7] = apd[11];
          const uint4* bp = (const uint4*)(W1t + ((size_t)(nt * KT1 + tk)) * 512 + lane * 16);
          b.q[0] = bp[0]; b.q[1] = bp[1];
          acc = wmma_bf16(a.v, b.v, acc);
        }
#pragma unroll
        for (int r = 0; r < 8; ++r)
          HF(mt * 16 + rowhi + r, pp * innerPad + nt * 16 + mrow) = acc[r];
      }
      __syncthreads();
      // 1b: bias + GELU + gate scale -> hs (packed at stride `inner`)
      int cnt = 32 * pc * inner;
      for (int idx = tid; idx < cnt; idx += 256) {
        int m = idx / (pc * inner);
        int c2 = idx - m * (pc * inner);
        int pp = c2 / inner;
        int i = c2 - pp * inner;
        float g = gates[(n0 + m) * N_EXPERT + eIdx];
        float vv = HF(m, pp * innerPad + i) + b1[i];
        HS(m, (p0 + pp) * inner + i) = f2bf(gelu_exact(vv) * g);
      }
      __syncthreads();
    }
    // tail: bias column at Ke, zeros up to Kp
    for (int idx = tid; idx < 32 * (Kp - Ke); idx += 256) {
      int m = idx / (Kp - Ke);
      int c = Ke + (idx - m * (Kp - Ke));
      float g = gates[(n0 + m) * N_EXPERT + eIdx];
      HS(m, c) = f2bf(c == Ke ? g : 0.0f);
    }
  } else {
    // VALU fallback (pl = 6, 3: ~0.12 GMAC total)
    for (int idx = tid; idx < 32 * Kp; idx += 256) {
      int m = idx / Kp, c = idx - m * Kp;
      float g = gates[(n0 + m) * N_EXPERT + eIdx];
      float hval;
      if (c < Ke) {
        int p = c / inner;
        int i = c - p * inner;
        float accv = b1[i];
        for (int j = 0; j < pl; ++j)
          accv += bf2f(XS(m, p * step + j)) * W1[j * inner + i];
        hval = gelu_exact(accv) * g;
      } else if (c == Ke) {
        hval = g;
      } else {
        hval = 0.0f;
      }
      HS(m, c) = f2bf(hval);
    }
  }
  __syncthreads();

  // Phase 2: WMMA GEMM, M=32 x N=512; wave w covers ntiles [4w, 4w+4)
  const int Kt = Kp >> 5;
  v8f acc[2][4];
#pragma unroll
  for (int mt = 0; mt < 2; ++mt)
#pragma unroll
    for (int q = 0; q < 4; ++q) acc[mt][q] = zero;

  for (int tk = 0; tk < Kt; ++tk) {
    Frag16 a0, a1;
    const uint4* ap0 = (const uint4*)&HS(mrow, tk * 32 + kh);
    a0.q[0] = ap0[0]; a0.q[1] = ap0[2];
    const uint4* ap1 = (const uint4*)&HS(16 + mrow, tk * 32 + kh);
    a1.q[0] = ap1[0]; a1.q[1] = ap1[2];
    // speculative prefetch of next B tile (past-end prefetch is dropped by HW)
    __builtin_prefetch(W2t + ((size_t)(w * 4 * Kt + tk + 1)) * 512 + lane * 16, 0, 1);
#pragma unroll
    for (int q = 0; q < 4; ++q) {
      int nt = w * 4 + q;
      Frag16 b;
      const uint4* bp = (const uint4*)(W2t + ((size_t)(nt * Kt + tk)) * 512 + lane * 16);
      b.q[0] = bp[0]; b.q[1] = bp[1];
      acc[0][q] = wmma_bf16(a0.v, b.v, acc[0][q]);
      acc[1][q] = wmma_bf16(a1.v, b.v, acc[1][q]);
    }
  }

#pragma unroll
  for (int mt = 0; mt < 2; ++mt) {
#pragma unroll
    for (int q = 0; q < 4; ++q) {
      int nt = w * 4 + q;
#pragma unroll
      for (int r = 0; r < 8; ++r) {
        size_t o = (n0 + mt * 16 + rowhi + r) * D_MODEL + nt * 16 + (lane & 15);
        float v = acc[mt][q][r];
        if (accumulate) v += out[o];
        out[o] = v;
      }
    }
  }
#undef XS
#undef HS
#undef HF
}

// ---------------------------------------------------------------------------
// Host launcher
// ---------------------------------------------------------------------------
extern "C" void kernel_launch(void* const* d_in, const int* in_sizes, int n_in,
                              void* d_out, int out_size, void* d_ws, size_t ws_size,
                              hipStream_t stream) {
  (void)in_sizes; (void)n_in; (void)out_size; (void)ws_size;

  // Expert configs
  static const int PL[6]    = {96, 48, 24, 12, 6, 3};
  static const int STEP[6]  = {48, 24, 12, 6, 3, 1};
  static const int PADL[6]  = {48, 24, 12, 6, 3, 1};
  static const int INNER[6] = {64, 34, 17, 8, 4, 1};
  static const int PNUM[6]  = {8, 15, 29, 57, 113, 337};
  static const int KE[6]    = {512, 510, 493, 456, 452, 337};
  static const int KP[6]    = {544, 512, 512, 480, 480, 352};
  // Patch-GEMM configs (layer 1): K1 = roundup32(pl), iPad = roundup16(inner)
  static const int PWMMA[6] = {1, 1, 1, 1, 0, 0};
  static const int KP1[6]   = {96, 64, 32, 32, 32, 32};
  static const int IPAD[6]  = {64, 48, 32, 16, 16, 16};
  static const int CHP[6]   = {8, 10, 16, 32, 1, 1};   // patches per Hf chunk

  const float* x = (const float*)d_in[0];
  const float* eW1[6]; const float* eB1[6]; const float* eW2[6]; const float* eB2[6];
  for (int e = 0; e < 6; ++e) {
    eW1[e] = (const float*)d_in[1 + 4 * e + 0];
    eB1[e] = (const float*)d_in[1 + 4 * e + 1];
    eW2[e] = (const float*)d_in[1 + 4 * e + 2];
    eB2[e] = (const float*)d_in[1 + 4 * e + 3];
  }
  const float* gW1 = (const float*)d_in[25];
  const float* gB1 = (const float*)d_in[26];
  const float* gW2 = (const float*)d_in[27];
  const float* gB2 = (const float*)d_in[28];
  float* out = (float*)d_out;

  // Workspace layout; Hg overlays Xbf, gates overlay RI (disjoint lifetimes)
  char* ws = (char*)d_ws;
  const size_t OFF_XBF = 0;
  const size_t OFF_BAS = OFF_XBF + (size_t)N_TOK * DFT_KP * 2;
  const size_t OFF_W1G = OFF_BAS + (size_t)DFT_KP * DFT_NC * 2;
  const size_t OFF_W2E = OFF_W1G + (size_t)RF_KP * GATE_H * 2;
  size_t w2eOff[7]; w2eOff[0] = OFF_W2E;
  for (int e = 0; e < 6; ++e) w2eOff[e + 1] = w2eOff[e] + (size_t)KP[e] * D_MODEL * 2;
  size_t w1eOff[7]; w1eOff[0] = w2eOff[6];
  for (int e = 0; e < 6; ++e) w1eOff[e + 1] = w1eOff[e] + (size_t)KP1[e] * IPAD[e] * 2;
  const size_t OFF_RI  = w1eOff[6];
  const size_t OFF_RF  = OFF_RI + (size_t)N_TOK * DFT_NC * 4;
  const size_t OFF_HG  = OFF_XBF;
  const size_t OFF_GT  = OFF_RI;

  unsigned short* Xbf  = (unsigned short*)(ws + OFF_XBF);
  unsigned short* BasT = (unsigned short*)(ws + OFF_BAS);
  unsigned short* W1gT = (unsigned short*)(ws + OFF_W1G);
  float*          RI   = (float*)(ws + OFF_RI);
  unsigned short* RF   = (unsigned short*)(ws + OFF_RF);
  float*          Hg   = (float*)(ws + OFF_HG);
  float*          GT   = (float*)(ws + OFF_GT);

  // 1) x -> bf16 (padded)
  k_x_to_bf16<<<(N_TOK * DFT_KP + 255) / 256, 256, 0, stream>>>(x, Xbf);

  // 2) pack DFT basis tiles
  k_pack_dft_tiles<<<(DFT_KP * DFT_NC + 255) / 256, 256, 0, stream>>>(BasT);

  // 3) pack gate W1 tiles (bias row = b1 at k=505)
  k_pack_b_tiles<<<(RF_KP * GATE_H + 255) / 256, 256, 0, stream>>>(
      gW1, gB1, RDIM, GATE_H, RDIM, RF_KP, GATE_H, W1gT);

  // 4) pack expert W2 tiles (bias row = b2 at k=Ke) and W1 tiles (layer 1)
  for (int e = 0; e < 6; ++e) {
    unsigned short* dst2 = (unsigned short*)(ws + w2eOff[e]);
    k_pack_b_tiles<<<(KP[e] * D_MODEL + 255) / 256, 256, 0, stream>>>(
        eW2[e], eB2[e], KE[e], D_MODEL, KE[e], KP[e], D_MODEL, dst2);
    if (PWMMA[e]) {
      unsigned short* dst1 = (unsigned short*)(ws + w1eOff[e]);
      k_pack_b_tiles<<<(KP1[e] * IPAD[e] + 255) / 256, 256, 0, stream>>>(
          eW1[e], (const float*)nullptr, PL[e], INNER[e], -1, KP1[e], IPAD[e], dst1);
    }
  }

  // 5) DFT GEMM: RI[N,352] = Xbf @ BasT
  {
    int tiles = (N_TOK / 16) * (DFT_NC / 16);   // 45056
    k_gemm_wave<<<tiles / 8, 256, 0, stream>>>(Xbf, BasT, RI, N_TOK, DFT_KP, DFT_NC);
  }

  // 6) rf assembly (x | magnitude | 1 | 0)
  k_build_rf<<<(N_TOK * RF_KP + 255) / 256, 256, 0, stream>>>(x, RI, RF);

  // 7) gate GEMM: Hg[N,128] = rf @ W1gT
  {
    int tiles = (N_TOK / 16) * (GATE_H / 16);   // 16384
    k_gemm_wave<<<tiles / 8, 256, 0, stream>>>(RF, W1gT, Hg, N_TOK, RF_KP, GATE_H);
  }

  // 8) top-4 softmax gates
  k_gate_topk<<<N_TOK / 256, 256, 0, stream>>>(Hg, gW2, gB2, GT);

  // 9) experts: fused patch-WMMA + GELU + WMMA GEMM
  for (int e = 0; e < 6; ++e) {
    const unsigned short* W2t = (const unsigned short*)(ws + w2eOff[e]);
    const unsigned short* W1t = (const unsigned short*)(ws + w1eOff[e]);
    k_expert_fused<<<N_TOK / 32, 256, SM_TOTAL, stream>>>(
        x, eW1[e], eB1[e], W1t, W2t, GT, out,
        PL[e], STEP[e], PADL[e], INNER[e], PNUM[e], KE[e], KP[e],
        PWMMA[e], KP1[e] / 32, IPAD[e] / 16, IPAD[e], CHP[e],
        e, e == 0 ? 0 : 1);
  }
}
